// W8A8LinearStatic_20169166422721
// MI455X (gfx1250) — 8-bit matrix operand order repaired and checked by reading, compiled, not run on hardware
//
#include <hip/hip_runtime.h>
#include <stdint.h>

typedef int      v8i __attribute__((ext_vector_type(8)));
typedef int      v4i __attribute__((ext_vector_type(4)));
typedef _Float16 v8h __attribute__((ext_vector_type(8)));

#define M_DIM 8192
#define K_DIM 4096
#define N_DIM 11008

#define TILE_M 128
#define TILE_N 256
#define TILE_K 64
#define LDSS   80

__global__ __launch_bounds__(256) void quant_fp16_to_i8(
    const _Float16* __restrict__ x, int8_t* __restrict__ xq,
    const float* __restrict__ input_scale, const int8_t* __restrict__ input_offset)
{
    const size_t t   = (size_t)blockIdx.x * 256 + threadIdx.x;
    const float  inv = 1.0f / input_scale[0];
    const int    off = (int)input_offset[0];

    const v8h* src = (const v8h*)(x + t * 16);
    v8h h0 = src[0];
    v8h h1 = src[1];

    alignas(16) int8_t q[16];
#pragma unroll
    for (int i = 0; i < 8; ++i) {
        int v = (int)__builtin_rintf((float)h0[i] * inv) + off;
        q[i] = (int8_t)(v < -128 ? -128 : (v > 127 ? 127 : v));
    }
#pragma unroll
    for (int i = 0; i < 8; ++i) {
        int v = (int)__builtin_rintf((float)h1[i] * inv) + off;
        q[8 + i] = (int8_t)(v < -128 ? -128 : (v > 127 ? 127 : v));
    }
    *(v4i*)(xq + t * 16) = *(const v4i*)q;
}

__device__ __forceinline__ void async_ld16(uint32_t lds_addr, uint32_t voff, const void* sbase)
{
    asm volatile("global_load_async_to_lds_b128 %0, %1, %2"
                 :: "v"(lds_addr), "v"(voff), "s"(sbase) : "memory");
}
__device__ __forceinline__ void wait_async_le6() { asm volatile("s_wait_asynccnt 0x6" ::: "memory"); }
__device__ __forceinline__ void wait_async_le0() { asm volatile("s_wait_asynccnt 0x0" ::: "memory"); }

__global__ __launch_bounds__(256) void w8a8_gemm_wmma(
    const int8_t* __restrict__ xq,
    const int8_t* __restrict__ w,
    const int*    __restrict__ qbias,
    const float*  __restrict__ dscale,
    _Float16*     __restrict__ out)
{
    __shared__ int8_t Alds[2][TILE_M * LDSS];
    __shared__ int8_t Blds[2][TILE_N * LDSS];

    const int tid  = threadIdx.x;
    const int lane = tid & 31;
    const int wid  = tid >> 5;
    const int wm   = wid >> 2;
    const int wn   = wid & 3;

    const int bm = blockIdx.y * TILE_M;
    const int bn = blockIdx.x * TILE_N;

    const int c4  = tid >> 2;
    const int c16 = (tid & 3) * 16;

    const int8_t* baseA = xq + (size_t)bm * K_DIM;
    const int8_t* baseB = w  + (size_t)bn * K_DIM;
    const uint32_t gcom0 = (uint32_t)(c4 * K_DIM + c16);

    const uint32_t ldsA0 = (uint32_t)(uintptr_t)&Alds[0][c4 * LDSS + c16];
    const uint32_t ldsA1 = (uint32_t)(uintptr_t)&Alds[1][c4 * LDSS + c16];
    const uint32_t ldsB0 = (uint32_t)(uintptr_t)&Blds[0][c4 * LDSS + c16];
    const uint32_t ldsB1 = (uint32_t)(uintptr_t)&Blds[1][c4 * LDSS + c16];

    v8i acc[4][4];
#pragma unroll
    for (int i = 0; i < 4; ++i)
#pragma unroll
        for (int j = 0; j < 4; ++j)
            acc[i][j] = (v8i){0, 0, 0, 0, 0, 0, 0, 0};

    const int fr    = lane & 15;
    const int aksel = (lane >> 4) * 16;
    const int bksel = (lane >> 4) * 16;

    {
        const uint32_t g = gcom0;
        async_ld16(ldsA0,                  g,                       baseA);
        async_ld16(ldsA0 + 64u * LDSS,     g + 64u * K_DIM,         baseA);
        async_ld16(ldsB0,                  g,                       baseB);
        async_ld16(ldsB0 + 64u * LDSS,     g + 64u * K_DIM,         baseB);
        async_ld16(ldsB0 + 128u * LDSS,    g + 128u * K_DIM,        baseB);
        async_ld16(ldsB0 + 192u * LDSS,    g + 192u * K_DIM,        baseB);
    }

    const int KT = K_DIM / TILE_K;
    for (int kt = 0; kt < KT; ++kt) {
        const int cur = kt & 1;

        if (kt + 1 < KT) {
            const uint32_t g  = gcom0 + (uint32_t)((kt + 1) * TILE_K);
            const uint32_t lA = cur ? ldsA0 : ldsA1;
            const uint32_t lB = cur ? ldsB0 : ldsB1;
            async_ld16(lA,                g,                baseA);
            async_ld16(lA + 64u * LDSS,   g + 64u * K_DIM,  baseA);
            async_ld16(lB,                g,                baseB);
            async_ld16(lB + 64u * LDSS,   g + 64u * K_DIM,  baseB);
            async_ld16(lB + 128u * LDSS,  g + 128u * K_DIM, baseB);
            async_ld16(lB + 192u * LDSS,  g + 192u * K_DIM, baseB);
            wait_async_le6();
        } else {
            wait_async_le0();
        }
        __syncthreads();

        const int8_t* Ab = &Alds[cur][0];
        const int8_t* Bb = &Blds[cur][0];

        v8i bf[4];
#pragma unroll
        for (int j = 0; j < 4; ++j) {
            const int8_t* p = Bb + (wn * 64 + j * 16 + fr) * LDSS + bksel;
            v4i q0 = *(const v4i*)p;
            v4i q1 = *(const v4i*)(p + 32);
            v8i b = {q0.x, q0.y, q0.z, q0.w, q1.x, q1.y, q1.z, q1.w};
            bf[j] = b;
        }

#pragma unroll
        for (int i = 0; i < 4; ++i) {
            const int8_t* p = Ab + (wm * 64 + i * 16 + fr) * LDSS + aksel;
            v4i p0 = *(const v4i*)p;
            v8i a;
            {
                const long2* q = (const long2*)0; (void)q;
            }
            const long* p64 = (const long*)p;
            long a0 = p64[0];
            long a1 = *(const long*)(p + 8);
            long a2 = *(const long*)(p + 32);
            long a3 = *(const long*)(p + 40);
            a = (v8i){(int)(a0), (int)(a0 >> 32), (int)(a1), (int)(a1 >> 32),
                      (int)(a2), (int)(a2 >> 32), (int)(a3), (int)(a3 >> 32)};
            (void)p0;
#pragma unroll
            for (int j = 0; j < 4; ++j)
                acc[i][j] = __builtin_amdgcn_wmma_i32_16x16x64_iu8(
                     true, a,  true, bf[j],
                    acc[i][j],  false,  false);
        }

        __syncthreads();
    }

    const int colbase = bn + wn * 64 + (lane & 15);
    const int row0    = bm + wm * 64 + ((lane >> 4) << 3);
#pragma unroll
    for (int j = 0; j < 4; ++j) {
        const int   col   = colbase + j * 16;
        const int   bias  = qbias[col];
        const float scale = dscale[col];
#pragma unroll
        for (int i = 0; i < 4; ++i) {
            const int r0 = row0 + i * 16;
#pragma unroll
            for (int r = 0; r < 8; ++r) {
                float f = (float)(acc[i][j][r] + bias) * scale;
                out[(size_t)(r0 + r) * N_DIM + col] = (_Float16)f;
            }
        }
    }
}

extern "C" void kernel_launch(void* const* d_in, const int* in_sizes, int n_in,
                              void* d_out, int out_size, void* d_ws, size_t ws_size,
                              hipStream_t stream)
{
    (void)in_sizes; (void)n_in; (void)out_size; (void)ws_size;

    const _Float16* x        = (const _Float16*)d_in[0];
    const int8_t*   weight   = (const int8_t*)  d_in[1];
    const float*    deqs     = (const float*)   d_in[2];
    const float*    in_scale = (const float*)   d_in[3];
    const int*      qbias    = (const int*)     d_in[4];
    const int8_t*   in_off   = (const int8_t*)  d_in[5];
    _Float16*       out      = (_Float16*)d_out;
    int8_t*         xq       = (int8_t*)d_ws;

    {
        const int threads = (M_DIM * K_DIM) / 16;
        quant_fp16_to_i8<<<threads / 256, 256, 0, stream>>>(x, xq, in_scale, in_off);
    }
    {
        dim3 grid(N_DIM / TILE_N, M_DIM / TILE_M);
        w8a8_gemm_wmma<<<grid, dim3(256, 1, 1), 0, stream>>>(xq, weight, qbias, deqs, out);
    }
}
